// TransformerDecoderLayer_74139725463948
// MI455X (gfx1250) — matrix operand order repaired and checked by reading, not compiled, not run on hardware
//
#include <hip/hip_runtime.h>
#include <hip/hip_bf16.h>
#include <math.h>

// ---------------- types ----------------
typedef __bf16 bf16;
typedef bf16  v16bf __attribute__((ext_vector_type(16)));
typedef float v8f   __attribute__((ext_vector_type(8)));
typedef unsigned int uint_t;
typedef uint_t v4u  __attribute__((ext_vector_type(4)));

#define D_MODEL 256
#define NHEAD   8
#define DKH     32
#define NL      500
#define NB      8
#define NROWS   (NL * NB)   // 4000

// swizzled-weight tile geometry (must match GEMM LDS image)
#define BM_T 128
#define BN_T 128
#define BK_T 32
#define CH_T 24                        // 16 frag elems + 8 pad -> 48B chunks
#define WTILE_ELEMS (2 * BN_T * CH_T)  // 6144 bf16 = 12288 B per (n_block, k_slice)

__device__ __forceinline__ unsigned short f2bf_bits(float f) {
    unsigned u = __builtin_bit_cast(unsigned, f);
    return (unsigned short)((u + 0x7FFFu + ((u >> 16) & 1u)) >> 16);  // RNE
}
__device__ __forceinline__ unsigned pack2bf(float lo, float hi) {
    return (unsigned)f2bf_bits(lo) | ((unsigned)f2bf_bits(hi) << 16);
}

__device__ __forceinline__ float wave_sum(float s) {
#pragma unroll
    for (int off = 16; off > 0; off >>= 1) s += __shfl_xor(s, off, 32);
    return s;
}

// ---------------- weight pre-swizzle: fp32 [K][N] -> bf16 LDS tile image ----------------
// blob layout: tile(nb, ks) at ((nb * K/32) + ks) * WTILE_ELEMS, element (n*2+h)*CH + e
__global__ __launch_bounds__(256)
void swizzle_w_kernel(const float* __restrict__ W, bf16* __restrict__ out, int K, int N)
{
    const int nb = blockIdx.x, ks = blockIdx.y;
    const int tid = threadIdx.x;
    bf16* dst = out + ((size_t)nb * (K >> 5) + ks) * WTILE_ELEMS;
#pragma unroll
    for (int i = 0; i < 8; ++i) {
        int p = tid + i * 256;          // 2048 K-pairs in a 32x128 tile
        int n = p & 127;
        int r = (p >> 7) * 2;           // even K index 0..30
        float w0 = W[(size_t)(ks * 32 + r)     * N + nb * 128 + n];
        float w1 = W[(size_t)(ks * 32 + r + 1) * N + nb * 128 + n];
        int h = (r >> 3) & 1;
        int e = (r & 7) + ((r >> 4) << 3);  // even
        *(uint_t*)&dst[(n * 2 + h) * CH_T + e] = pack2bf(w0, w1);
    }
}

// ---------------- WMMA GEMM: C[M,N] = act( (A (+A2)) @ W + bias ) ----------------
// Block tile 128x128, BK=32, 256 threads (8 waves, 4x2), wave tile 32x64
// -> 8 v_wmma_f32_16x16x32_bf16 per K-step per wave.
// A tile: fp32 global -> bf16 LDS, pre-swizzled to fragment order (packed b32 stores).
// B tile: raw 12KB copy of pre-swizzled bf16 weights via global_load_async_to_lds_b128.
template<bool RELU, bool FUSE_ADD>
__global__ __launch_bounds__(256)
void gemm_wmma_kernel(const float* __restrict__ A, const float* __restrict__ A2,
                      const bf16* __restrict__ Wsw, const float* __restrict__ bias,
                      float* __restrict__ C, int M, int N, int K)
{
    constexpr int BM = BM_T, BN = BN_T, BK = BK_T, CH = CH_T;
    __shared__ bf16 AsP[2 * BM * CH];   // chunk index = (m*2 + hi)
    __shared__ bf16 BsP[2 * BN * CH];   // chunk index = (n*2 + hi)

    const int tid  = threadIdx.x;
    const int wave = tid >> 5;
    const int lane = tid & 31;
    const int wm   = wave & 3;
    const int wn   = wave >> 2;
    const int block_m = blockIdx.y * BM;
    const int block_n = blockIdx.x * BN;
    const int fm = lane & 15;
    const int hi = lane >> 4;

    v8f acc[2][4] = {};

    const unsigned lds_b_base = (unsigned)(size_t)&BsP[0];   // low 32 bits = LDS offset

    for (int k0 = 0; k0 < K; k0 += BK) {
        // ---- B tile: async raw copy of pre-swizzled weights (768 x 16B) ----
        {
            const bf16* wsrc = Wsw + ((size_t)blockIdx.x * (K >> 5) + (k0 >> 5)) * WTILE_ELEMS;
#pragma unroll
            for (int i = 0; i < 3; ++i) {
                int t = tid + i * 256;
                unsigned lofs = lds_b_base + (unsigned)t * 16u;
                unsigned long long g = (unsigned long long)(size_t)wsrc + (size_t)t * 16u;
                asm volatile("global_load_async_to_lds_b128 %0, %1, off"
                             :: "v"(lofs), "v"(g) : "memory");
            }
        }
        // prefetch next A tile while async copy runs
        if (k0 + BK < K) {
            int pr = min(block_m + (tid >> 1), M - 1);
            __builtin_prefetch(&A[(size_t)pr * K + k0 + BK + (tid & 1) * 16], 0, 3);
        }
        // ---- A tile (128x32): 8 float2 loads -> 8 packed b32 LDS stores / thread ----
#pragma unroll
        for (int i = 0; i < 8; ++i) {
            int p  = tid + i * 256;
            int r  = p >> 4;
            int c  = (p & 15) * 2;
            int gr = min(block_m + r, M - 1);
            float2 v = *(const float2*)&A[(size_t)gr * K + k0 + c];
            if (FUSE_ADD) {
                float2 v2 = *(const float2*)&A2[(size_t)gr * K + k0 + c];
                v.x += v2.x; v.y += v2.y;
            }
            int h = (c >> 3) & 1;
            int e = (c & 7) + ((c >> 4) << 3);
            *(uint_t*)&AsP[(r * 2 + h) * CH + e] = pack2bf(v.x, v.y);
        }
        asm volatile("s_wait_asynccnt 0" ::: "memory");   // drain async B copy
        __syncthreads();

        // ---- fragment loads: 2x ds_load_b128 per fragment ----
        union FC { struct { v4u lo, hi2; } u; v16bf f; };
        FC a0, a1, b[4];
        {
            const bf16* pa0 = &AsP[((wm * 32 + fm) * 2 + hi) * CH];
            const bf16* pa1 = &AsP[((wm * 32 + 16 + fm) * 2 + hi) * CH];
            a0.u.lo = *(const v4u*)pa0;  a0.u.hi2 = *(const v4u*)(pa0 + 8);
            a1.u.lo = *(const v4u*)pa1;  a1.u.hi2 = *(const v4u*)(pa1 + 8);
#pragma unroll
            for (int ni = 0; ni < 4; ++ni) {
                const bf16* pb = &BsP[((wn * 64 + ni * 16 + fm) * 2 + hi) * CH];
                b[ni].u.lo = *(const v4u*)pb;  b[ni].u.hi2 = *(const v4u*)(pb + 8);
            }
        }
#pragma unroll
        for (int ni = 0; ni < 4; ++ni) {
            acc[0][ni] = __builtin_amdgcn_wmma_f32_16x16x32_bf16(false, a0.f, false, b[ni].f, (short)0, acc[0][ni], false, false);
            acc[1][ni] = __builtin_amdgcn_wmma_f32_16x16x32_bf16(false, a1.f, false, b[ni].f, (short)0, acc[1][ni], false, false);
        }
        __syncthreads();
    }

    // ---- epilogue: full blocks take the unguarded path ----
    if (block_m + BM <= M) {
#pragma unroll
        for (int mi = 0; mi < 2; ++mi)
#pragma unroll
            for (int ni = 0; ni < 4; ++ni)
#pragma unroll
                for (int vv = 0; vv < 8; ++vv) {
                    int gr = block_m + wm * 32 + mi * 16 + vv + 8 * hi;
                    int gc = block_n + wn * 64 + ni * 16 + fm;
                    float val = acc[mi][ni][vv] + bias[gc];
                    if (RELU) val = fmaxf(val, 0.f);
                    C[(size_t)gr * N + gc] = val;
                }
    } else {
#pragma unroll
        for (int mi = 0; mi < 2; ++mi)
#pragma unroll
            for (int ni = 0; ni < 4; ++ni)
#pragma unroll
                for (int vv = 0; vv < 8; ++vv) {
                    int gr = block_m + wm * 32 + mi * 16 + vv + 8 * hi;
                    int gc = block_n + wn * 64 + ni * 16 + fm;
                    if (gr < M) {
                        float val = acc[mi][ni][vv] + bias[gc];
                        if (RELU) val = fmaxf(val, 0.f);
                        C[(size_t)gr * N + gc] = val;
                    }
                }
    }
}

// ---------------- self-attention (flash-style, wave per query) ----------------
__global__ __launch_bounds__(256)
void attn_kernel(const float* __restrict__ qkv, float* __restrict__ ctx)
{
    __shared__ float Ks[64][33];
    __shared__ float Vs[64][33];
    const int hb   = blockIdx.x;     // (b<<3)|h
    const int b    = hb >> 3;
    const int h    = hb & 7;
    const int tid  = threadIdx.x;
    const int wave = tid >> 5;
    const int lane = tid & 31;
    const int q_idx = blockIdx.y * 8 + wave;

    float qv = 0.f;
    if (q_idx < NL) qv = qkv[((size_t)q_idx * NB + b) * 768 + h * 32 + lane];

    float m_run = -1e30f, l_run = 0.f, acc = 0.f;
    const float scale = 0.17677669529663687f;  // 1/sqrt(32)

    for (int t0 = 0; t0 < NL; t0 += 64) {
#pragma unroll
        for (int i = 0; i < 8; ++i) {
            int idx = tid + i * 256;
            int ml = idx >> 5, dd = idx & 31;
            int m = t0 + ml;
            float kv = 0.f, vv = 0.f;
            if (m < NL) {
                size_t rb = ((size_t)m * NB + b) * 768 + h * 32 + dd;
                kv = qkv[rb + 256];
                vv = qkv[rb + 512];
            }
            Ks[ml][dd] = kv;
            Vs[ml][dd] = vv;
        }
        __syncthreads();
        if (q_idx < NL) {
            int nvalid = min(64, NL - t0);
            for (int ml = 0; ml < nvalid; ++ml) {
                float s = wave_sum(qv * Ks[ml][lane]) * scale;
                float m_new = fmaxf(m_run, s);
                float corr = __expf(m_run - m_new);
                float p    = __expf(s - m_new);
                l_run = l_run * corr + p;
                acc   = acc * corr + p * Vs[ml][lane];
                m_run = m_new;
            }
        }
        __syncthreads();
    }
    if (q_idx < NL)
        ctx[((size_t)q_idx * NB + b) * 256 + h * 32 + lane] = acc / l_run;
}

// ---------------- multi-scale deformable sampling ----------------
__global__ __launch_bounds__(256)
void deform_kernel(const float* __restrict__ offs, const float* __restrict__ attl,
                   const float* __restrict__ refp,
                   const float* __restrict__ f0, const float* __restrict__ f1,
                   const float* __restrict__ f2, const float* __restrict__ f3,
                   float* __restrict__ out)
{
    const int r = blockIdx.x;            // r = l*NB + b
    const int b = r & (NB - 1);
    const int tid = threadIdx.x;
    const int h = tid >> 5;
    const int d = tid & 31;

    const float rx = refp[(size_t)r * 2 + 0];
    const float ry = refp[(size_t)r * 2 + 1];

    float lg[16];
    float mx = -1e30f;
#pragma unroll
    for (int i = 0; i < 16; ++i) { lg[i] = attl[(size_t)r * 128 + h * 16 + i]; mx = fmaxf(mx, lg[i]); }
    float sum = 0.f;
#pragma unroll
    for (int i = 0; i < 16; ++i) { lg[i] = __expf(lg[i] - mx); sum += lg[i]; }
    const float inv = 1.f / sum;

    const float* feats[4] = {f0, f1, f2, f3};
    const int    HS[4]    = {128, 64, 32, 16};
    const int    c        = h * 32 + d;

    float acc = 0.f;
#pragma unroll
    for (int s = 0; s < 4; ++s) {
        const int Hl = HS[s], Wl = HS[s];
        const float* fs = feats[s];
#pragma unroll
        for (int k = 0; k < 4; ++k) {
            float ox = offs[(size_t)r * 256 + h * 32 + s * 8 + k * 2 + 0];
            float oy = offs[(size_t)r * 256 + h * 32 + s * 8 + k * 2 + 1];
            float px = (rx + ox / (float)Wl) * (float)(Wl - 1);
            float py = (ry + oy / (float)Hl) * (float)(Hl - 1);
            float valid = (px >= 0.f && px <= (float)(Wl - 1) &&
                           py >= 0.f && py <= (float)(Hl - 1)) ? 1.f : 0.f;
            float fx = floorf(px), fy = floorf(py);
            float wx = px - fx, wy = py - fy;
            int x0 = min(max((int)fx, 0), Wl - 1);
            int x1 = min(x0 + 1, Wl - 1);
            int y0 = min(max((int)fy, 0), Hl - 1);
            int y1 = min(y0 + 1, Hl - 1);
            size_t base = (size_t)b * Hl * Wl;
            float v00 = fs[(base + (size_t)y0 * Wl + x0) * 256 + c];
            float v01 = fs[(base + (size_t)y0 * Wl + x1) * 256 + c];
            float v10 = fs[(base + (size_t)y1 * Wl + x0) * 256 + c];
            float v11 = fs[(base + (size_t)y1 * Wl + x1) * 256 + c];
            float bil = v00 * (1.f - wy) * (1.f - wx) + v01 * (1.f - wy) * wx
                      + v10 * wy * (1.f - wx)        + v11 * wy * wx;
            acc += lg[s * 4 + k] * inv * bil * valid;
        }
    }
    out[(size_t)r * 256 + c] = acc;
}

// ---------------- residual + LayerNorm (wave per row) ----------------
__global__ __launch_bounds__(256)
void ln_residual_kernel(const float* __restrict__ a, const float* __restrict__ bres,
                        const float* __restrict__ g, const float* __restrict__ be,
                        float* __restrict__ out, int rows)
{
    const int tid  = threadIdx.x;
    const int wave = tid >> 5;
    const int lane = tid & 31;
    const int row  = blockIdx.x * 8 + wave;
    if (row >= rows) return;
    float v[8];
    float s = 0.f;
#pragma unroll
    for (int i = 0; i < 8; ++i) {
        int c = lane + i * 32;
        v[i] = a[(size_t)row * 256 + c] + bres[(size_t)row * 256 + c];
        s += v[i];
    }
    s = wave_sum(s);
    const float mean = s * (1.f / 256.f);
    float var = 0.f;
#pragma unroll
    for (int i = 0; i < 8; ++i) { float d = v[i] - mean; var += d * d; }
    var = wave_sum(var) * (1.f / 256.f);
    const float rs = rsqrtf(var + 1e-5f);
#pragma unroll
    for (int i = 0; i < 8; ++i) {
        int c = lane + i * 32;
        out[(size_t)row * 256 + c] = (v[i] - mean) * rs * g[c] + be[c];
    }
}

// ---------------- host orchestration ----------------
static inline void launch_gemm(const float* A, const float* A2, const bf16* Wsw, const float* bias,
                               float* C, int M, int N, int K, bool relu, bool fuse,
                               hipStream_t stream)
{
    dim3 grid(N / 128, (M + 127) / 128);
    if (fuse)       gemm_wmma_kernel<false, true ><<<grid, 256, 0, stream>>>(A, A2, Wsw, bias, C, M, N, K);
    else if (relu)  gemm_wmma_kernel<true,  false><<<grid, 256, 0, stream>>>(A, A2, Wsw, bias, C, M, N, K);
    else            gemm_wmma_kernel<false, false><<<grid, 256, 0, stream>>>(A, A2, Wsw, bias, C, M, N, K);
}

extern "C" void kernel_launch(void* const* d_in, const int* in_sizes, int n_in,
                              void* d_out, int out_size, void* d_ws, size_t ws_size,
                              hipStream_t stream)
{
    (void)in_sizes; (void)n_in; (void)out_size; (void)ws_size;

    const float* tgt  = (const float*)d_in[0];
    const float* refp = (const float*)d_in[1];
    const float* mem[4] = {(const float*)d_in[2], (const float*)d_in[4],
                           (const float*)d_in[6], (const float*)d_in[8]};
    const float* pos[4] = {(const float*)d_in[3], (const float*)d_in[5],
                           (const float*)d_in[7], (const float*)d_in[9]};
    const float* Wqkv = (const float*)d_in[10]; const float* bqkv = (const float*)d_in[11];
    const float* Wo   = (const float*)d_in[12]; const float* bo   = (const float*)d_in[13];
    const float* Woff = (const float*)d_in[14]; const float* boff = (const float*)d_in[15];
    const float* Watt = (const float*)d_in[16]; const float* batt = (const float*)d_in[17];
    const float* Wk   = (const float*)d_in[18]; const float* bk   = (const float*)d_in[19];
    const float* Wm   = (const float*)d_in[20]; const float* bm   = (const float*)d_in[21];
    const float* W1   = (const float*)d_in[22]; const float* b1   = (const float*)d_in[23];
    const float* W2   = (const float*)d_in[24]; const float* b2   = (const float*)d_in[25];
    const float* g1   = (const float*)d_in[26]; const float* be1  = (const float*)d_in[27];
    const float* g2   = (const float*)d_in[28]; const float* be2  = (const float*)d_in[29];
    const float* g3   = (const float*)d_in[30]; const float* be3  = (const float*)d_in[31];

    float* ws = (float*)d_ws;
    size_t o = 0;
    float* qkv  = ws + o; o += (size_t)NROWS * 768;
    float* ctx  = ws + o; o += (size_t)NROWS * 256;
    float* sa   = ws + o; o += (size_t)NROWS * 256;
    float* x1   = ws + o; o += (size_t)NROWS * 256;
    float* offs = ws + o; o += (size_t)NROWS * 256;
    float* attl = ws + o; o += (size_t)NROWS * 128;
    float* accb = ws + o; o += (size_t)NROWS * 256;
    float* ca   = ws + o; o += (size_t)NROWS * 256;
    float* x2   = ws + o; o += (size_t)NROWS * 256;
    float* ffn1 = ws + o; o += (size_t)NROWS * 2048;
    float* ffn2 = ws + o; o += (size_t)NROWS * 256;
    float* feat[4];
    const int HWs[4] = {128, 64, 32, 16};
    for (int s = 0; s < 4; ++s) { feat[s] = ws + o; o += (size_t)NB * HWs[s] * HWs[s] * 256; }

    // carve bf16 swizzled-weight blobs (element count = N/128 * K/32 * WTILE_ELEMS)
    auto carve = [&](int K, int N) {
        bf16* p = (bf16*)(ws + o);
        size_t elems = (size_t)(N / 128) * (size_t)(K / 32) * WTILE_ELEMS;
        o += (elems + 1) / 2;
        return p;
    };
    bf16* WqkvS = carve(256, 768);
    bf16* WoS   = carve(256, 256);
    bf16* WoffS = carve(256, 256);
    bf16* WattS = carve(256, 128);
    bf16* WkS   = carve(256, 256);
    bf16* WmS   = carve(256, 256);
    bf16* W1S   = carve(256, 2048);
    bf16* W2S   = carve(2048, 256);

    // 0. pre-swizzle all weight matrices to bf16 tile images
    swizzle_w_kernel<<<dim3(768 / 128, 8),  256, 0, stream>>>(Wqkv, WqkvS, 256, 768);
    swizzle_w_kernel<<<dim3(2, 8),          256, 0, stream>>>(Wo,   WoS,   256, 256);
    swizzle_w_kernel<<<dim3(2, 8),          256, 0, stream>>>(Woff, WoffS, 256, 256);
    swizzle_w_kernel<<<dim3(1, 8),          256, 0, stream>>>(Watt, WattS, 256, 128);
    swizzle_w_kernel<<<dim3(2, 8),          256, 0, stream>>>(Wk,   WkS,   256, 256);
    swizzle_w_kernel<<<dim3(2, 8),          256, 0, stream>>>(Wm,   WmS,   256, 256);
    swizzle_w_kernel<<<dim3(16, 8),         256, 0, stream>>>(W1,   W1S,   256, 2048);
    swizzle_w_kernel<<<dim3(2, 64),         256, 0, stream>>>(W2,   W2S,   2048, 256);

    // 1. QKV projection
    launch_gemm(tgt, nullptr, WqkvS, bqkv, qkv, NROWS, 768, 256, false, false, stream);
    // 2. self-attention
    attn_kernel<<<dim3(NB * NHEAD, (NL + 7) / 8), 256, 0, stream>>>(qkv, ctx);
    // 3. output projection
    launch_gemm(ctx, nullptr, WoS, bo, sa, NROWS, 256, 256, false, false, stream);
    // 4. x1 = LN(tgt + sa)
    ln_residual_kernel<<<(NROWS + 7) / 8, 256, 0, stream>>>(tgt, sa, g1, be1, x1, NROWS);
    // 5. sampling offsets + attention logits
    launch_gemm(x1, nullptr, WoffS, boff, offs, NROWS, 256, 256, false, false, stream);
    launch_gemm(x1, nullptr, WattS, batt, attl, NROWS, 128, 256, false, false, stream);
    // 6. per-scale value projection: feat_s = (memory_s + poses_s) @ Wk + bk
    for (int s = 0; s < 4; ++s)
        launch_gemm(mem[s], pos[s], WkS, bk, feat[s],
                    NB * HWs[s] * HWs[s], 256, 256, false, true, stream);
    // 7. deformable sampling
    deform_kernel<<<NROWS, 256, 0, stream>>>(offs, attl, refp,
                                             feat[0], feat[1], feat[2], feat[3], accb);
    // 8. Wm projection + LN2
    launch_gemm(accb, nullptr, WmS, bm, ca, NROWS, 256, 256, false, false, stream);
    ln_residual_kernel<<<(NROWS + 7) / 8, 256, 0, stream>>>(x1, ca, g2, be2, x2, NROWS);
    // 9. FFN
    launch_gemm(x2, nullptr, W1S, b1, ffn1, NROWS, 2048, 256, true, false, stream);
    launch_gemm(ffn1, nullptr, W2S, b2, ffn2, NROWS, 256, 2048, false, false, stream);
    // 10. final LN -> d_out
    ln_residual_kernel<<<(NROWS + 7) / 8, 256, 0, stream>>>(x2, ffn2, g3, be3,
                                                            (float*)d_out, NROWS);
}
